// MultiHeadAttention_35837207117915
// MI455X (gfx1250) — compile-verified, matrix operand order checked by reading, not run on hardware
//
#include <hip/hip_runtime.h>
#include <hip/hip_bf16.h>

typedef __attribute__((ext_vector_type(16))) _Float16 v16h;
typedef __attribute__((ext_vector_type(4)))  _Float16 v4h;
typedef __attribute__((ext_vector_type(8)))  float    v8f;
typedef __attribute__((ext_vector_type(8)))  unsigned int v8u;

#define EMBED  768
#define NHEADS 12
#define HDIM   64
#define BATCH  4
#define SEQ    2048
#define ROWS   (BATCH * SEQ)   // 8192

// K index for dword-pair p (p=0..7) within a 16-bit A/B fragment, per ISA
// 7.12.2: lanes 0-15 hold K {0..7,16..23}, lanes 16-31 hold K {8..15,24..31}.
__device__ __forceinline__ int kpair(int p, int hl) {
    return (p < 4 ? 2 * p : 2 * p + 8) + hl * 8;
}

__device__ __forceinline__ v16h load_frag(const _Float16* base, int hl) {
    v8u u;
#pragma unroll
    for (int p = 0; p < 8; ++p)
        u[p] = *(const unsigned int*)&base[kpair(p, hl)];
    return __builtin_bit_cast(v16h, u);
}

// ---------------------------------------------------------------------------
// fp32 -> fp16 conversion, 4-wide (all our sizes are multiples of 4)
// ---------------------------------------------------------------------------
__global__ void cvt_f32_f16_v4(const float4* __restrict__ src,
                               v4h* __restrict__ dst, int n4) {
    int i = blockIdx.x * blockDim.x + threadIdx.x;
    int stride = gridDim.x * blockDim.x;
    for (; i < n4; i += stride) {
        float4 f = src[i];
        v4h h;
        h[0] = (_Float16)f.x; h[1] = (_Float16)f.y;
        h[2] = (_Float16)f.z; h[3] = (_Float16)f.w;
        dst[i] = h;
    }
}

// ---------------------------------------------------------------------------
// Y[row, col] = sum_k A[row,k] * W[col,k] + bias[col]
// A: [M x K] f16 row-major, W: [Ntot x K] f16 row-major (torch Linear W)
// mode 0: scatter to Qh/Kh/Vh as [B,H,N,hd] halves   (QKV projection)
// mode 1: store f32 to outf [M x Ntot]               (output projection)
// Block: 256 threads = 8 waves. Tile 128x64, K-step 32, double-buffered LDS.
// Each wave owns a 2x2 grid of 16x16 subtiles: 4 WMMA per 4 fragment loads.
// ---------------------------------------------------------------------------
__global__ __launch_bounds__(256)
void gemm_wmma(const _Float16* __restrict__ A,
               const _Float16* __restrict__ W,
               const float*    __restrict__ bias,
               int M, int Ntot, int K,
               _Float16* __restrict__ qh, _Float16* __restrict__ kh,
               _Float16* __restrict__ vh,
               float* __restrict__ outf, int mode) {
    __shared__ __align__(16) _Float16 At[2][128 * 32];
    __shared__ __align__(16) _Float16 Bt[2][64 * 32];

    const int tid  = threadIdx.x;
    const int lane = tid & 31;
    const int wave = tid >> 5;       // 0..7
    const int hl   = lane >> 4;      // lane-half (K offset select)
    const int ln   = lane & 15;      // N / M index within fragment

    const int mblk = blockIdx.x * 128;
    const int nblk = blockIdx.y * 64;

    const int mg = wave & 3;         // M subtile group (rows mg*32 .. +31)
    const int ng = wave >> 2;        // N subtile group (cols ng*32 .. +31)

    v8f acc[2][2];
#pragma unroll
    for (int i = 0; i < 2; ++i)
#pragma unroll
        for (int j = 0; j < 2; ++j) acc[i][j] = (v8f){0, 0, 0, 0, 0, 0, 0, 0};

    // cooperative staging: A 128x32 (16 halves/thread), B 64x32 (8/thread)
    const int arow = tid >> 1;           // 0..127
    const int acol = (tid & 1) * 16;     // 0 or 16
    const int brow = tid >> 2;           // 0..63
    const int bcol = (tid & 3) * 8;      // 0,8,16,24

    // prologue: stage K-slice 0 into buffer 0
    *(uint4*)&At[0][arow * 32 + acol] =
        *(const uint4*)&A[(size_t)(mblk + arow) * K + acol];
    *(uint4*)&At[0][arow * 32 + acol + 8] =
        *(const uint4*)&A[(size_t)(mblk + arow) * K + acol + 8];
    *(uint4*)&Bt[0][brow * 32 + bcol] =
        *(const uint4*)&W[(size_t)(nblk + brow) * K + bcol];
    __syncthreads();

    const int nk = K / 32;
    for (int ks = 0; ks < nk; ++ks) {
        const int buf = ks & 1;
        const bool pre = (ks + 1 < nk);

        uint4 ra0, ra1, rb0;
        if (pre) {
            const int kb2 = (ks + 1) * 32;
            ra0 = *(const uint4*)&A[(size_t)(mblk + arow) * K + kb2 + acol];
            ra1 = *(const uint4*)&A[(size_t)(mblk + arow) * K + kb2 + acol + 8];
            rb0 = *(const uint4*)&W[(size_t)(nblk + brow) * K + kb2 + bcol];
        }

        v16h af[2], bfv[2];
#pragma unroll
        for (int m2 = 0; m2 < 2; ++m2)
            af[m2] = load_frag(&At[buf][((mg * 2 + m2) * 16 + ln) * 32], hl);
#pragma unroll
        for (int nj = 0; nj < 2; ++nj)
            bfv[nj] = load_frag(&Bt[buf][((ng * 2 + nj) * 16 + ln) * 32], hl);

#pragma unroll
        for (int m2 = 0; m2 < 2; ++m2)
#pragma unroll
            for (int nj = 0; nj < 2; ++nj)
                acc[m2][nj] = __builtin_amdgcn_wmma_f32_16x16x32_f16(
                    false, af[m2], false, bfv[nj], (short)0, acc[m2][nj],
                    false, false);

        if (pre) {
            *(uint4*)&At[buf ^ 1][arow * 32 + acol]     = ra0;
            *(uint4*)&At[buf ^ 1][arow * 32 + acol + 8] = ra1;
            *(uint4*)&Bt[buf ^ 1][brow * 32 + bcol]     = rb0;
        }
        __syncthreads();
    }

    // epilogue
#pragma unroll
    for (int m2 = 0; m2 < 2; ++m2) {
#pragma unroll
        for (int nj = 0; nj < 2; ++nj) {
            const int col = nblk + (ng * 2 + nj) * 16 + ln;   // fixed per lane
            const float bv = bias[col];
#pragma unroll
            for (int r = 0; r < 8; ++r) {
                const int m   = r + 8 * hl;
                const int row = mblk + (mg * 2 + m2) * 16 + m;
                const float v = acc[m2][nj][r] + bv;
                if (mode == 0) {
                    const int which = col / EMBED;
                    const int rem   = col % EMBED;
                    const int h = rem / HDIM, d = rem % HDIM;
                    const int b = row / SEQ,  n = row % SEQ;
                    const size_t idx =
                        (((size_t)(b * NHEADS + h)) * SEQ + n) * HDIM + d;
                    const _Float16 hv = (_Float16)v;
                    if (which == 0)      qh[idx] = hv;
                    else if (which == 1) kh[idx] = hv;
                    else                 vh[idx] = hv;
                } else {
                    outf[(size_t)row * Ntot + col] = v;
                }
            }
        }
    }
}

// ---------------------------------------------------------------------------
// Flash attention: block = 128 threads = 4 waves; each wave owns 32 queries
// (two 16-row subtiles) and the full 64-dim output accumulation, so K/V
// fragments are reused across both query subtiles. Key/value tiles of 64 are
// double-buffered in LDS. Per-wave P restaging uses same-wave in-order LDS
// semantics (DScnt), so no block barrier is needed between P write and read.
// grid = (SEQ/128, B*H)
// ---------------------------------------------------------------------------
__global__ __launch_bounds__(128)
void flash_attn(const _Float16* __restrict__ Qh, const _Float16* __restrict__ Kh,
                const _Float16* __restrict__ Vh, _Float16* __restrict__ Oh) {
    __shared__ __align__(16) _Float16 Kt[2][64 * 64];      // [key][dim]
    __shared__ __align__(16) _Float16 Vt[2][64 * 64];      // [dim][key]
    __shared__ __align__(16) _Float16 Pb[4 * 32 * 64];     // per-wave P tiles

    const int tid  = threadIdx.x;
    const int lane = tid & 31;
    const int wave = tid >> 5;       // 0..3
    const int hl   = lane >> 4;
    const int ln   = lane & 15;

    const int bh    = blockIdx.y;                    // b*H + h
    const int qbase = blockIdx.x * 128 + wave * 32;  // 32 queries per wave

    const _Float16* Q  = Qh + (size_t)bh * SEQ * HDIM;
    const _Float16* Kp = Kh + (size_t)bh * SEQ * HDIM;
    const _Float16* Vp = Vh + (size_t)bh * SEQ * HDIM;

    // Q fragments: 2 query subtiles x (K = 0..31, 32..63), kept live.
    v16h qf[2][2];
#pragma unroll
    for (int m2 = 0; m2 < 2; ++m2)
#pragma unroll
        for (int f = 0; f < 2; ++f) {
            v8u u;
#pragma unroll
            for (int p = 0; p < 8; ++p)
                u[p] = *(const unsigned int*)
                    &Q[(size_t)(qbase + m2 * 16 + ln) * HDIM + f * 32 +
                       kpair(p, hl)];
            qf[m2][f] = __builtin_bit_cast(v16h, u);
        }

    const float scale = 0.125f;   // 1/sqrt(64)

    float rmax[2][8], rsum[2][8];
    v8f   oacc[2][4];
#pragma unroll
    for (int m2 = 0; m2 < 2; ++m2) {
#pragma unroll
        for (int r = 0; r < 8; ++r) { rmax[m2][r] = -3.0e30f; rsum[m2][r] = 0.0f; }
#pragma unroll
        for (int i = 0; i < 4; ++i) oacc[m2][i] = (v8f){0, 0, 0, 0, 0, 0, 0, 0};
    }

    // staging geometry: thread t handles key row t>>1, dim half (t&1)*32
    const int srow = tid >> 1;
    const int scol = (tid & 1) * 32;

    // prologue: stage key tile 0 into buffer 0
    {
#pragma unroll
        for (int q = 0; q < 4; ++q)
            *(uint4*)&Kt[0][srow * 64 + scol + q * 8] =
                *(const uint4*)&Kp[(size_t)srow * HDIM + scol + q * 8];
        uint4 rv[4];
#pragma unroll
        for (int q = 0; q < 4; ++q)
            rv[q] = *(const uint4*)&Vp[(size_t)srow * HDIM + scol + q * 8];
        const unsigned int* rw = (const unsigned int*)rv;
#pragma unroll
        for (int e = 0; e < 16; ++e) {
            const unsigned int w = rw[e];
            const int d0 = scol + 2 * e;
            Vt[0][d0 * 64 + srow] =
                __builtin_bit_cast(_Float16, (unsigned short)(w & 0xffff));
            Vt[0][(d0 + 1) * 64 + srow] =
                __builtin_bit_cast(_Float16, (unsigned short)(w >> 16));
        }
    }
    __syncthreads();

    const int ntiles = SEQ / 64;
    for (int t = 0; t < ntiles; ++t) {
        const int buf = t & 1;
        const bool pre = (t + 1 < ntiles);

        uint4 rk[4], rv[4];
        if (pre) {
            const int kt2 = (t + 1) * 64;
#pragma unroll
            for (int q = 0; q < 4; ++q) {
                rk[q] = *(const uint4*)
                    &Kp[(size_t)(kt2 + srow) * HDIM + scol + q * 8];
                rv[q] = *(const uint4*)
                    &Vp[(size_t)(kt2 + srow) * HDIM + scol + q * 8];
            }
        }

        // S = Q * K^T : 2 query subtiles x 4 key subtiles; K b-frags shared.
        v8f sacc[2][4];
#pragma unroll
        for (int m2 = 0; m2 < 2; ++m2)
#pragma unroll
            for (int ni = 0; ni < 4; ++ni)
                sacc[m2][ni] = (v8f){0, 0, 0, 0, 0, 0, 0, 0};
#pragma unroll
        for (int ni = 0; ni < 4; ++ni) {
#pragma unroll
            for (int f = 0; f < 2; ++f) {
                v16h bf = load_frag(
                    &Kt[buf][(ni * 16 + ln) * 64 + f * 32], hl);
#pragma unroll
                for (int m2 = 0; m2 < 2; ++m2)
                    sacc[m2][ni] = __builtin_amdgcn_wmma_f32_16x16x32_f16(
                        false, qf[m2][f], false, bf, (short)0, sacc[m2][ni],
                        false, false);
            }
        }

        // Online softmax per query subtile. Row r of the C/D layout lives in
        // vgpr r across the 16 lanes of this lane-half.
#pragma unroll
        for (int m2 = 0; m2 < 2; ++m2) {
            float mnew[8], alpha[8];
#pragma unroll
            for (int r = 0; r < 8; ++r) {
                float mx = rmax[m2][r];
#pragma unroll
                for (int ni = 0; ni < 4; ++ni) {
                    const float v = sacc[m2][ni][r] * scale;
                    sacc[m2][ni][r] = v;
                    mx = fmaxf(mx, v);
                }
                mx = fmaxf(mx, __shfl_xor(mx, 1));
                mx = fmaxf(mx, __shfl_xor(mx, 2));
                mx = fmaxf(mx, __shfl_xor(mx, 4));
                mx = fmaxf(mx, __shfl_xor(mx, 8));
                mnew[r]  = mx;
                alpha[r] = __expf(rmax[m2][r] - mx);
                rmax[m2][r] = mx;
            }
#pragma unroll
            for (int r = 0; r < 8; ++r) {
                float psum = 0.0f;
#pragma unroll
                for (int ni = 0; ni < 4; ++ni) {
                    const float p = __expf(sacc[m2][ni][r] - mnew[r]);
                    psum += p;
                    Pb[wave * 2048 + (m2 * 16 + r + 8 * hl) * 64 + ni * 16 +
                       ln] = (_Float16)p;
                }
                psum += __shfl_xor(psum, 1);
                psum += __shfl_xor(psum, 2);
                psum += __shfl_xor(psum, 4);
                psum += __shfl_xor(psum, 8);
                rsum[m2][r] = rsum[m2][r] * alpha[r] + psum;
#pragma unroll
                for (int ni = 0; ni < 4; ++ni) oacc[m2][ni][r] *= alpha[r];
            }
        }

        // O += P * V : P a-frags per subtile, V b-frags shared across subtiles.
        v16h pf[2][2];
#pragma unroll
        for (int m2 = 0; m2 < 2; ++m2)
#pragma unroll
            for (int f = 0; f < 2; ++f)
                pf[m2][f] = load_frag(
                    &Pb[wave * 2048 + (m2 * 16 + ln) * 64 + f * 32], hl);
#pragma unroll
        for (int ni = 0; ni < 4; ++ni) {
#pragma unroll
            for (int f = 0; f < 2; ++f) {
                v16h vf = load_frag(
                    &Vt[buf][(ni * 16 + ln) * 64 + f * 32], hl);
#pragma unroll
                for (int m2 = 0; m2 < 2; ++m2)
                    oacc[m2][ni] = __builtin_amdgcn_wmma_f32_16x16x32_f16(
                        false, pf[m2][f], false, vf, (short)0, oacc[m2][ni],
                        false, false);
            }
        }

        if (pre) {
#pragma unroll
            for (int q = 0; q < 4; ++q)
                *(uint4*)&Kt[buf ^ 1][srow * 64 + scol + q * 8] = rk[q];
            const unsigned int* rw = (const unsigned int*)rv;
#pragma unroll
            for (int e = 0; e < 16; ++e) {
                const unsigned int w = rw[e];
                const int d0 = scol + 2 * e;
                Vt[buf ^ 1][d0 * 64 + srow] =
                    __builtin_bit_cast(_Float16, (unsigned short)(w & 0xffff));
                Vt[buf ^ 1][(d0 + 1) * 64 + srow] =
                    __builtin_bit_cast(_Float16, (unsigned short)(w >> 16));
            }
        }
        __syncthreads();
    }

    // Finalize: O /= rowsum, write halves to [B, N, C] for the proj GEMM.
    const int b = bh / NHEADS, h = bh % NHEADS;
#pragma unroll
    for (int m2 = 0; m2 < 2; ++m2)
#pragma unroll
        for (int ni = 0; ni < 4; ++ni)
#pragma unroll
            for (int r = 0; r < 8; ++r) {
                const int m = m2 * 16 + r + 8 * hl;
                const int d = ni * 16 + ln;
                const float v = oacc[m2][ni][r] / rsum[m2][r];
                Oh[((size_t)(b * SEQ + qbase + m)) * EMBED + h * HDIM + d] =
                    (_Float16)v;
            }
}

// ---------------------------------------------------------------------------
extern "C" void kernel_launch(void* const* d_in, const int* in_sizes, int n_in,
                              void* d_out, int out_size, void* d_ws,
                              size_t ws_size, hipStream_t stream) {
    const float* x      = (const float*)d_in[0];
    const float* qkv_w  = (const float*)d_in[1];
    const float* qkv_b  = (const float*)d_in[2];
    const float* proj_w = (const float*)d_in[3];
    const float* proj_b = (const float*)d_in[4];
    float* out = (float*)d_out;

    char* ws = (char*)d_ws;
    size_t off = 0;
    auto wsalloc = [&](size_t bytes) -> void* {
        void* p = ws + off;
        off = (off + bytes + 255) & ~(size_t)255;
        return p;
    };

    _Float16* Xh     = (_Float16*)wsalloc((size_t)ROWS * EMBED * 2);
    _Float16* Wqkvh  = (_Float16*)wsalloc((size_t)3 * EMBED * EMBED * 2);
    _Float16* Wprojh = (_Float16*)wsalloc((size_t)EMBED * EMBED * 2);
    _Float16* Qh     = (_Float16*)wsalloc((size_t)ROWS * HDIM * NHEADS * 2);
    _Float16* Kh     = (_Float16*)wsalloc((size_t)ROWS * HDIM * NHEADS * 2);
    _Float16* Vh     = (_Float16*)wsalloc((size_t)ROWS * HDIM * NHEADS * 2);
    _Float16* AttnH  = (_Float16*)wsalloc((size_t)ROWS * EMBED * 2);

    // fp32 -> fp16 conversions (4-wide)
    cvt_f32_f16_v4<<<1024, 256, 0, stream>>>((const float4*)x, (v4h*)Xh,
                                             ROWS * EMBED / 4);
    cvt_f32_f16_v4<<<1024, 256, 0, stream>>>((const float4*)qkv_w,
                                             (v4h*)Wqkvh,
                                             3 * EMBED * EMBED / 4);
    cvt_f32_f16_v4<<<512, 256, 0, stream>>>((const float4*)proj_w,
                                            (v4h*)Wprojh, EMBED * EMBED / 4);

    // QKV projection: [8192 x 768] @ [2304 x 768]^T -> Q/K/V [B,H,N,hd]
    gemm_wmma<<<dim3(ROWS / 128, (3 * EMBED) / 64), 256, 0, stream>>>(
        Xh, Wqkvh, qkv_b, ROWS, 3 * EMBED, EMBED, Qh, Kh, Vh, nullptr, 0);

    // Flash attention -> AttnH [B, N, C] halves
    flash_attn<<<dim3(SEQ / 128, BATCH * NHEADS), 128, 0, stream>>>(Qh, Kh, Vh,
                                                                    AttnH);

    // Output projection: [8192 x 768] @ [768 x 768]^T + b -> out (f32)
    gemm_wmma<<<dim3(ROWS / 128, EMBED / 64), 256, 0, stream>>>(
        AttnH, Wprojh, proj_b, ROWS, EMBED, EMBED, nullptr, nullptr, nullptr,
        out, 1);
}
